// PointNet_44985487458409
// MI455X (gfx1250) — matrix operand order repaired and checked by reading, not compiled, not run on hardware
//
#include <hip/hip_runtime.h>
#include <hip/hip_fp16.h>

typedef __attribute__((ext_vector_type(16))) _Float16 v16h;
typedef __attribute__((ext_vector_type(8)))  _Float16 v8h;
typedef __attribute__((ext_vector_type(8)))  float    v8f;

#define Bn    8
#define Cc    39
#define Nn    8192
#define Mm    2048
#define Kk    32
#define NCOLI 524288   /* Bn*Mm*Kk */
#define R2f   9.0f

// ---------------------------------------------------------------------------
// Ball query: one thread per (b,m). First-in-index-order neighbors, pad first.
// ---------------------------------------------------------------------------
__global__ void bq_k(const float* __restrict__ x, int* __restrict__ idxb) {
  int t = blockIdx.x * blockDim.x + threadIdx.x;
  if (t >= Bn * Mm) return;
  int m = t % Mm, b = t / Mm;
  const float* x0 = x + (size_t)b * (Cc * Nn);
  const float* x1 = x0 + Nn;
  const float* x2 = x0 + 2 * Nn;
  float qx = x0[4 * m], qy = x1[4 * m], qz = x2[4 * m];
  int* ip = idxb + (size_t)t * Kk;
  int cnt = 0, first = 0;
  for (int j = 0; j < Nn; ++j) {
    float dx = x0[j] - qx, dy = x1[j] - qy, dz = x2[j] - qz;
    float d2 = fmaf(dx, dx, fmaf(dy, dy, dz * dz));
    if (d2 <= R2f) {
      if (cnt == 0) first = j;
      ip[cnt] = j;
      if (++cnt == Kk) break;
    }
  }
  for (int k = cnt; k < Kk; ++k) ip[k] = first;
}

// ---------------------------------------------------------------------------
// Gather -> v0t [NCOL][64] f16 (channels contiguous per point, rows 38..63 = 0)
// ---------------------------------------------------------------------------
__global__ void gather_k(const float* __restrict__ x,
                         const int* __restrict__ idxb,
                         _Float16* __restrict__ v0) {
  int col = blockIdx.x * blockDim.x + threadIdx.x;
  if (col >= NCOLI) return;
  const int j  = idxb[col];
  const int bm = col / Kk;
  const int m  = bm % Mm;
  const int b  = bm / Mm;
  const float* xb = x + (size_t)b * (Cc * Nn);
  _Float16 tmp[64];
#pragma unroll
  for (int c2 = 0; c2 < 3; ++c2)
    tmp[c2] = (_Float16)(xb[c2 * Nn + j] - xb[c2 * Nn + 4 * m]);
#pragma unroll
  for (int c2 = 3; c2 < 6; ++c2)
    tmp[c2] = (_Float16)xb[c2 * Nn + j];
#pragma unroll
  for (int c2 = 6; c2 < 38; ++c2)
    tmp[c2] = (_Float16)xb[(c2 + 1) * Nn + j];
#pragma unroll
  for (int c2 = 38; c2 < 64; ++c2)
    tmp[c2] = (_Float16)0.f;
  v8h* dst = (v8h*)(v0 + (size_t)col * 64);
#pragma unroll
  for (int q = 0; q < 8; ++q) {
    v8h hh;
#pragma unroll
    for (int i = 0; i < 8; ++i) hh[i] = tmp[q * 8 + i];
    dst[q] = hh;
  }
}

// Convert / pad weights fp32 -> f16 (zero pad cin), row-major [cout][cind].
__global__ void wconv_k(const float* __restrict__ W, _Float16* __restrict__ Wh,
                        int cout, int cins, int cind) {
  int t = blockIdx.x * blockDim.x + threadIdx.x;
  if (t >= cout * cind) return;
  int c = t % cind, o = t / cind;
  Wh[t] = (c < cins) ? (_Float16)W[(size_t)o * cins + c] : (_Float16)0.f;
}

__global__ void zero_k(float* p, int n) {
  int t = blockIdx.x * blockDim.x + threadIdx.x;
  if (t < n) p[t] = 0.f;
}

// ---------------------------------------------------------------------------
// WMMA GEMM:  Yt[NCOL][COUT] = Xt[NCOL][CIN] * W[COUT][CIN]^T  (f16 in/out,
// fp32 accumulate) + per-output-channel sum/sumsq for BN.
// Block = 128 threads = 4 waves on one 16-row cout strip (blockIdx.y).
// Each wave processes TWO independent 16-column tiles per iteration:
// all B loads are issued up front, and the two WMMA chains are independent,
// so loads of one tile overlap WMMAs of the other.
// ---------------------------------------------------------------------------
template <int CIN, int COUT, int NITER>
__global__ __launch_bounds__(128) void gemm_f16(
    const _Float16* __restrict__ Wh, const _Float16* __restrict__ X,
    _Float16* __restrict__ Y, float* __restrict__ osum,
    float* __restrict__ osq) {
  constexpr int KS = CIN / 32;
  const int tid  = threadIdx.x;
  const int lane = tid & 31;
  const int wave = tid >> 5;
  const int half = lane >> 4;
  const int l16  = lane & 15;
  const int m0   = blockIdx.y * 16;

  __shared__ float ls[16], lq[16];
  if (tid < 16) { ls[tid] = 0.f; lq[tid] = 0.f; }
  __syncthreads();

  // A 16x32 f16 layout: h[0..7] -> K=half*8+i, h[8..15] -> K=16+half*8+i
  v16h afrag[KS];
#pragma unroll
  for (int ks = 0; ks < KS; ++ks) {
    const _Float16* wr = Wh + (size_t)(m0 + l16) * CIN + ks * 32 + half * 8;
    v8h lo = *(const v8h*)(wr);
    v8h hi = *(const v8h*)(wr + 16);
#pragma unroll
    for (int i = 0; i < 8; ++i) { afrag[ks][i] = lo[i]; afrag[ks][8 + i] = hi[i]; }
  }

  float rs[8], rq[8];
#pragma unroll
  for (int r = 0; r < 8; ++r) { rs[r] = 0.f; rq[r] = 0.f; }

  // 8 tiles per block-iteration: wave w owns tiles (w*2, w*2+1).
  const int tile0 = blockIdx.x * (8 * NITER) + wave * 2;
  const _Float16* xp = X + ((size_t)tile0 * 16 + l16) * CIN;
  _Float16*       yp = Y + ((size_t)tile0 * 16 + l16) * COUT + m0 + half * 8;
  const size_t xstep = (size_t)128 * CIN;    // 8 tiles of 16 cols per iter
  const size_t ystep = (size_t)128 * COUT;
  const size_t xt1   = (size_t)16 * CIN;     // second tile offset
  const size_t yt1   = (size_t)16 * COUT;

  for (int it = 0; it < NITER; ++it) {
    // Issue ALL B loads for both tiles first (B 32x16 layout: h[0..7] ->
    // K=half*8+i, h[8..15] -> K=16+half*8+i; channels contiguous).
    v8h l0[KS], h0[KS], l1[KS], h1[KS];
#pragma unroll
    for (int ks = 0; ks < KS; ++ks) {
      const _Float16* p0 = xp + ks * 32 + half * 8;
      l0[ks] = *(const v8h*)(p0);
      h0[ks] = *(const v8h*)(p0 + 16);
      l1[ks] = *(const v8h*)(p0 + xt1);
      h1[ks] = *(const v8h*)(p0 + xt1 + 16);
    }
    v8f acc0 = {}, acc1 = {};
#pragma unroll
    for (int ks = 0; ks < KS; ++ks) {
      v16h bf;
#pragma unroll
      for (int i = 0; i < 8; ++i) { bf[i] = l0[ks][i]; bf[8 + i] = h0[ks][i]; }
      acc0 = __builtin_amdgcn_wmma_f32_16x16x32_f16(
          false, afrag[ks], false, bf, (short)0, acc0, false, false);
    }
#pragma unroll
    for (int ks = 0; ks < KS; ++ks) {
      v16h bf;
#pragma unroll
      for (int i = 0; i < 8; ++i) { bf[i] = l1[ks][i]; bf[8 + i] = h1[ks][i]; }
      acc1 = __builtin_amdgcn_wmma_f32_16x16x32_f16(
          false, afrag[ks], false, bf, (short)0, acc1, false, false);
    }
    v8h oh0, oh1;
#pragma unroll
    for (int r = 0; r < 8; ++r) {
      const float v0 = acc0[r], v1 = acc1[r];
      oh0[r] = (_Float16)v0;
      oh1[r] = (_Float16)v1;
      rs[r] += v0 + v1;
      rq[r] += v0 * v0 + v1 * v1;
    }
    *(v8h*)(yp)       = oh0;
    *(v8h*)(yp + yt1) = oh1;
    xp += xstep;
    yp += ystep;
  }

  // Reduce 16 lanes of each half-wave, then LDS, then one atomic per channel.
#pragma unroll
  for (int r = 0; r < 8; ++r) {
    float s = rs[r], q = rq[r];
#pragma unroll
    for (int msk = 8; msk >= 1; msk >>= 1) {
      s += __shfl_xor(s, msk, 32);
      q += __shfl_xor(q, msk, 32);
    }
    if (l16 == 0) {
      atomicAdd(&ls[half * 8 + r], s);
      atomicAdd(&lq[half * 8 + r], q);
    }
  }
  __syncthreads();
  if (tid < 16) {
    atomicAdd(&osum[m0 + tid], ls[tid]);
    atomicAdd(&osq[m0 + tid], lq[tid]);
  }
}

// BN coefficients: a = g*rsqrt(var+eps), c = b - mean*a
__global__ void coef_k(const float* __restrict__ s, const float* __restrict__ q,
                       const float* __restrict__ g, const float* __restrict__ b,
                       float* __restrict__ a, float* __restrict__ c, int cout) {
  int o = blockIdx.x * blockDim.x + threadIdx.x;
  if (o >= cout) return;
  const float invn = 1.0f / (float)NCOLI;
  float mean = s[o] * invn;
  float var  = q[o] * invn - mean * mean;
  float av   = g[o] * rsqrtf(var + 1e-5f);
  a[o] = av;
  c[o] = b[o] - mean * av;
}

// In-place BN+ReLU over Yt[NCOL][CIN]: 8 consecutive channels per thread.
__global__ void norm_k(_Float16* __restrict__ y, const float* __restrict__ a,
                       const float* __restrict__ c, int cinMask, int total8) {
  int t = blockIdx.x * blockDim.x + threadIdx.x;
  if (t >= total8) return;
  const size_t base = (size_t)t * 8;
  const int ch0 = (int)(base & (size_t)cinMask);
  v8h h = *(const v8h*)(y + base);
  const float4 a0 = *(const float4*)(a + ch0);
  const float4 a1 = *(const float4*)(a + ch0 + 4);
  const float4 c0 = *(const float4*)(c + ch0);
  const float4 c1 = *(const float4*)(c + ch0 + 4);
  const float av[8] = {a0.x, a0.y, a0.z, a0.w, a1.x, a1.y, a1.z, a1.w};
  const float cv[8] = {c0.x, c0.y, c0.z, c0.w, c1.x, c1.y, c1.z, c1.w};
  v8h o;
#pragma unroll
  for (int i = 0; i < 8; ++i)
    o[i] = (_Float16)fmaxf(fmaf((float)h[i], av[i], cv[i]), 0.f);
  *(v8h*)(y + base) = o;
}

// Output channels 0..6 = downsampled first 7 input channels.
__global__ void pcopy_k(const float* __restrict__ x, float* __restrict__ out) {
  int t = blockIdx.x * blockDim.x + threadIdx.x;
  if (t >= Bn * 7 * Mm) return;
  int m = t % Mm;
  int c = (t / Mm) % 7;
  int b = t / (7 * Mm);
  out[(size_t)b * (263 * Mm) + (size_t)c * Mm + m] =
      x[(size_t)b * (Cc * Nn) + (size_t)c * Nn + 4 * m];
}

// Channels 7..262: max over K of relu(a3*y3+c3).  Block per (b,m); thread = o.
// Reads 256 consecutive halves per k-row -> fully coalesced.
__global__ void maxout_k(const _Float16* __restrict__ y3t,
                         const float* __restrict__ a3,
                         const float* __restrict__ c3,
                         float* __restrict__ out) {
  const int bm = blockIdx.x;            // 0..B*M-1
  const int o  = threadIdx.x;           // 0..255
  const int m  = bm % Mm;
  const int b  = bm / Mm;
  const float a = a3[o], c = c3[o];
  const _Float16* yp = y3t + (size_t)bm * (Kk * 256) + o;
  float mx = -3.0e38f;
#pragma unroll
  for (int k = 0; k < Kk; ++k)
    mx = fmaxf(mx, fmaf((float)yp[(size_t)k * 256], a, c));
  mx = fmaxf(mx, 0.f);   // relu is monotone: max(relu) == relu(max)
  out[(size_t)b * (263 * Mm) + (size_t)(7 + o) * Mm + m] = mx;
}

// ---------------------------------------------------------------------------
extern "C" void kernel_launch(void* const* d_in, const int* in_sizes, int n_in,
                              void* d_out, int out_size, void* d_ws, size_t ws_size,
                              hipStream_t stream) {
  const float* x  = (const float*)d_in[0];
  const float* W1 = (const float*)d_in[1];
  const float* g1 = (const float*)d_in[2];
  const float* b1 = (const float*)d_in[3];
  const float* W2 = (const float*)d_in[4];
  const float* g2 = (const float*)d_in[5];
  const float* b2 = (const float*)d_in[6];
  const float* W3 = (const float*)d_in[7];
  const float* g3 = (const float*)d_in[8];
  const float* b3 = (const float*)d_in[9];
  float* out = (float*)d_out;

  char* w = (char*)d_ws;
  size_t off = 0;
  int* idxb = (int*)(w + off);            off += (size_t)NCOLI * 4;
  _Float16* v0t = (_Float16*)(w + off);   off += (size_t)NCOLI * 64 * 2;
  _Float16* y1t = (_Float16*)(w + off);   off += (size_t)NCOLI * 64 * 2;
  _Float16* y2t = (_Float16*)(w + off);   off += (size_t)NCOLI * 128 * 2;
  _Float16* y3t = (_Float16*)(w + off);   off += (size_t)NCOLI * 256 * 2;
  _Float16* wh1 = (_Float16*)(w + off);   off += 64  * 64  * 2;
  _Float16* wh2 = (_Float16*)(w + off);   off += 128 * 64  * 2;
  _Float16* wh3 = (_Float16*)(w + off);   off += 256 * 128 * 2;
  float* stats = (float*)(w + off);       off += 896 * 4;   // s1,q1,s2,q2,s3,q3
  float* coefs = (float*)(w + off);       off += 896 * 4;   // a1,c1,a2,c2,a3,c3
  float* s1 = stats, *q1 = s1 + 64, *s2 = q1 + 64, *q2 = s2 + 128,
       * s3 = q2 + 128, *q3 = s3 + 256;
  float* a1 = coefs, *c1 = a1 + 64, *a2 = c1 + 64, *c2 = a2 + 128,
       * a3 = c2 + 128, *c3 = a3 + 256;

  // stats must start at zero every call (atomically accumulated)
  zero_k<<<4, 256, 0, stream>>>(stats, 896);
  // weights -> f16 (zero-padded cin for layer 1: 38 -> 64)
  wconv_k<<<16, 256, 0, stream>>>(W1, wh1, 64, 38, 64);
  wconv_k<<<32, 256, 0, stream>>>(W2, wh2, 128, 64, 64);
  wconv_k<<<128, 256, 0, stream>>>(W3, wh3, 256, 128, 128);

  bq_k<<<64, 256, 0, stream>>>(x, idxb);
  gather_k<<<NCOLI / 256, 256, 0, stream>>>(x, idxb, v0t);

  // Layer 1: 64 x (64-padded 38)
  gemm_f16<64, 64, 8><<<dim3(512, 4), 128, 0, stream>>>(wh1, v0t, y1t, s1, q1);
  coef_k<<<1, 64, 0, stream>>>(s1, q1, g1, b1, a1, c1, 64);
  norm_k<<<NCOLI * 64 / 8 / 256, 256, 0, stream>>>(y1t, a1, c1, 63,
                                                   NCOLI * 64 / 8);

  // Layer 2: 128 x 64
  gemm_f16<64, 128, 8><<<dim3(512, 8), 128, 0, stream>>>(wh2, y1t, y2t, s2, q2);
  coef_k<<<1, 128, 0, stream>>>(s2, q2, g2, b2, a2, c2, 128);
  norm_k<<<NCOLI * 128 / 8 / 256, 256, 0, stream>>>(y2t, a2, c2, 127,
                                                    NCOLI * 128 / 8);

  // Layer 3: 256 x 128 (BN3+ReLU fused into maxout)
  gemm_f16<128, 256, 8><<<dim3(512, 16), 128, 0, stream>>>(wh3, y2t, y3t, s3, q3);
  coef_k<<<1, 256, 0, stream>>>(s3, q3, g3, b3, a3, c3, 256);

  // Output assembly
  pcopy_k<<<(Bn * 7 * Mm + 255) / 256, 256, 0, stream>>>(x, out);
  maxout_k<<<Bn * Mm, 256, 0, stream>>>(y3t, a3, c3, out);
}
